// NeuroSAT_13013750907359
// MI455X (gfx1250) — compile-verified, matrix operand order checked by reading, not run on hardware
//
#include <hip/hip_runtime.h>

#define FM     256
#define NVARS  4000
#define NLITS  8000
#define NCLS   16800
#define NEDGE  (NCLS * 3)
#define ROUNDS 32

typedef __attribute__((ext_vector_type(16))) _Float16 v16h;
typedef __attribute__((ext_vector_type(8)))  float    v8f;

__device__ __forceinline__ v8f wmma_f16(v16h a, v16h b, v8f c) {
  // D = A(16x32 f16) * B(32x16 f16) + C(16x16 f32)
  return __builtin_amdgcn_wmma_f32_16x16x32_f16(false, a, false, b, (short)0, c,
                                                false, false);
}

__device__ __forceinline__ void wait_async0() {
#if __has_builtin(__builtin_amdgcn_s_wait_asynccnt)
  __builtin_amdgcn_s_wait_asynccnt(0);
#else
  asm volatile("s_wait_asynccnt 0x0" ::: "memory");
#endif
}

// ---------------------------------------------------------------------------
// Weight packing: W is [Kd x N] f32 row-major. Output: per (32x16) K-N tile,
// 32 lanes x 16 f16 contiguous, matching the 16-bit B-fragment VGPR layout
// (lane = (n, half); elems j<8 -> K = half*8+j, j>=8 -> K = 16+half*8+(j-8)).
// ---------------------------------------------------------------------------
__global__ void pack_w_k(const float* __restrict__ W, _Float16* __restrict__ P,
                         int Kd, int N) {
  int tid = blockIdx.x * blockDim.x + threadIdx.x;
  int total = (Kd >> 5) * (N >> 4) * 32;
  if (tid >= total) return;
  int lane = tid & 31;
  int t = tid >> 5;
  int ntiles = N >> 4;
  int tn = t % ntiles, tk = t / ntiles;
  int n = tn * 16 + (lane & 15);
  int half = lane >> 4;
  int kb = tk * 32 + half * 8;
  _Float16* out = P + (size_t)tid * 16;
#pragma unroll
  for (int j = 0; j < 8; ++j) {
    out[j]     = (_Float16)W[(size_t)(kb + j) * N + n];
    out[8 + j] = (_Float16)W[(size_t)(kb + 16 + j) * N + n];
  }
}

// ---------------------------------------------------------------------------
// WMMA GEMM: C[M,N] = act(A[M,Kd] @ W + bias + prev). Each wave: 16x64 tile.
// Block: 256 threads = 8 waves = 128 rows. Grid: (N/64, ceil(M/128)).
// B tiles (4KB per k-step, shared by all 8 waves) are staged global->LDS with
// async copies, double-buffered; fragments then read via ds_load.
// ---------------------------------------------------------------------------
__global__ __launch_bounds__(256) void gemm_wmma_k(
    const float* __restrict__ A, const _Float16* __restrict__ Bp,
    const float* __restrict__ bias, const float* __restrict__ prev,
    float* __restrict__ C, int M, int N, int Kd, int relu) {
  __shared__ __align__(32) _Float16 sB[2][4 * 512];  // 2 bufs x 4 tiles x 1KB

  const int tid = threadIdx.x;
  const int lane = tid & 31;
  const int wave = tid >> 5;
  const int half = lane >> 4;
  const int nbase = blockIdx.x * 64;
  const int mbase = blockIdx.y * 128 + wave * 16;
  const bool active = (mbase < M);  // wave-uniform; inactive waves still barrier
  const int mrow = mbase + (lane & 15);
  const int ntiles = N >> 4;
  const int tn0 = nbase >> 4;
  const int kts = Kd >> 5;

  // Async-stage the contiguous 4KB of packed B for k-tile kt into buffer buf:
  // 256 threads x 16B each -> one global_load_async_to_lds_b128 per thread.
  auto stage = [&](int kt, int buf) {
    unsigned long long src =
        (unsigned long long)(const void*)(Bp + ((size_t)kt * ntiles + tn0) * 512) +
        (unsigned long long)(tid * 16);
    unsigned lp = (unsigned)(uintptr_t)(&sB[buf][0]) + (unsigned)(tid * 16);
    asm volatile("global_load_async_to_lds_b128 %0, %1, off"
                 :: "v"(lp), "v"(src)
                 : "memory");
  };

  v8f acc0 = {}, acc1 = {}, acc2 = {}, acc3 = {};

  stage(0, 0);
  wait_async0();
  __syncthreads();

  for (int kt = 0; kt < kts; ++kt) {
    const int buf = kt & 1;
    if (kt + 1 < kts) stage(kt + 1, buf ^ 1);

    if (active) {
      // A fragment: lane (m, half); two contiguous 8-float chunks -> f16
      const float* arow = A + (size_t)mrow * Kd + kt * 32 + half * 8;
      if (kt + 1 < kts) __builtin_prefetch(arow + 32, 0, 1);
      v16h af;
#pragma unroll
      for (int j = 0; j < 8; ++j) {
        af[j]     = (_Float16)arow[j];
        af[j + 8] = (_Float16)arow[16 + j];
      }
      const v16h b0 = *(const v16h*)&sB[buf][0 * 512 + lane * 16];
      const v16h b1 = *(const v16h*)&sB[buf][1 * 512 + lane * 16];
      const v16h b2 = *(const v16h*)&sB[buf][2 * 512 + lane * 16];
      const v16h b3 = *(const v16h*)&sB[buf][3 * 512 + lane * 16];
      acc0 = wmma_f16(af, b0, acc0);
      acc1 = wmma_f16(af, b1, acc1);
      acc2 = wmma_f16(af, b2, acc2);
      acc3 = wmma_f16(af, b3, acc3);
    }

    if (kt + 1 < kts) wait_async0();  // own async copies for kt+1 landed
    __syncthreads();                  // everyone staged + done reading buf
  }

  if (active) {
    const int ncol = lane & 15;
#pragma unroll
    for (int nt = 0; nt < 4; ++nt) {
      v8f acc = nt == 0 ? acc0 : nt == 1 ? acc1 : nt == 2 ? acc2 : acc3;
      int col = nbase + nt * 16 + ncol;
      float bv = bias ? bias[col] : 0.0f;
#pragma unroll
      for (int v = 0; v < 8; ++v) {
        int row = mbase + half * 8 + v;
        float x = acc[v] + bv;
        if (prev) x += prev[(size_t)row * N + col];
        if (relu) x = fmaxf(x, 0.0f);
        C[(size_t)row * N + col] = x;
      }
    }
  }
}

// ---------------------------------------------------------------------------
// Edge preprocessing from signed 1-based literals.
// ---------------------------------------------------------------------------
__global__ void edges_k(const int* __restrict__ lits, int* __restrict__ litIdx,
                        int* __restrict__ varIdx, float* __restrict__ sgn) {
  int e = blockIdx.x * blockDim.x + threadIdx.x;
  if (e >= NEDGE) return;
  int L = lits[e];
  int v = (L > 0 ? L : -L) - 1;
  varIdx[e] = v;
  sgn[e] = (L > 0) ? 1.0f : -1.0f;
  litIdx[e] = (L > 0) ? v : (NVARS + v);
}

__global__ void init_state_k(float* __restrict__ h, float* __restrict__ c,
                             const float* __restrict__ initRow, int M) {
  int idx = blockIdx.x * blockDim.x + threadIdx.x;
  if (idx >= M * FM) return;
  h[idx] = initRow[idx & (FM - 1)] * 0.0625f;  // / sqrt(256)
  c[idx] = 0.0f;
}

__global__ void fill_k(float* __restrict__ p, float v, int n) {
  int i = blockIdx.x * blockDim.x + threadIdx.x;
  if (i < n) p[i] = v;
}

// lc_msgs[c] = sum_{j<3} pre[litIdx[3c+j]]   (clause segments are sorted)
__global__ void gather3_k(const float* __restrict__ pre,
                          const int* __restrict__ litIdx,
                          float* __restrict__ msgs) {
  int idx = blockIdx.x * blockDim.x + threadIdx.x;
  if (idx >= NCLS * (FM / 4)) return;
  int c = idx >> 6;
  int q = (idx & 63) * 4;
  int r0 = litIdx[3 * c], r1 = litIdx[3 * c + 1], r2 = litIdx[3 * c + 2];
  const float* p0 = pre + (size_t)r0 * FM + q;
  const float* p1 = pre + (size_t)r1 * FM + q;
  const float* p2 = pre + (size_t)r2 * FM + q;
  float* d = msgs + (size_t)c * FM + q;
#pragma unroll
  for (int t = 0; t < 4; ++t) d[t] = p0[t] + p1[t] + p2[t];
}

// cl_msgs[litIdx[e]] += pre[clause(e)]   (unsorted segments -> atomics)
__global__ void scatter_add_k(const float* __restrict__ pre,
                              const int* __restrict__ litIdx,
                              float* __restrict__ clm) {
  int idx = blockIdx.x * blockDim.x + threadIdx.x;
  if (idx >= NEDGE * (FM / 4)) return;
  int e = idx >> 6;
  int q = (idx & 63) * 4;
  int c = e / 3;
  int l = litIdx[e];
  const float* src = pre + (size_t)c * FM + q;
  float* dst = clm + (size_t)l * FM + q;
#pragma unroll
  for (int t = 0; t < 4; ++t) atomicAdd(dst + t, src[t]);
}

// x = concat([cl_msgs, lh rows rotated by NVARS], axis=1)  -> [NLITS, 512]
__global__ void build_xcat_k(const float* __restrict__ clm,
                             const float* __restrict__ lh,
                             float* __restrict__ x) {
  int idx = blockIdx.x * blockDim.x + threadIdx.x;
  if (idx >= NLITS * 2 * FM) return;
  int row = idx >> 9;
  int col = idx & 511;
  float v;
  if (col < FM) {
    v = clm[(size_t)row * FM + col];
  } else {
    int fr = (row < NVARS) ? (row + NVARS) : (row - NVARS);
    v = lh[(size_t)fr * FM + (col - FM)];
  }
  x[idx] = v;
}

// vote input: concat([lh[:NVARS], lh[NVARS:]], axis=1) -> [NVARS, 512]
__global__ void build_vcat_k(const float* __restrict__ lh, float* __restrict__ x) {
  int idx = blockIdx.x * blockDim.x + threadIdx.x;
  if (idx >= NVARS * 2 * FM) return;
  int row = idx >> 9;
  int col = idx & 511;
  float v = (col < FM) ? lh[(size_t)row * FM + col]
                       : lh[(size_t)(row + NVARS) * FM + (col - FM)];
  x[idx] = v;
}

__device__ __forceinline__ float sigmoidf_(float x) {
  return 1.0f / (1.0f + expf(-x));
}

// Keras LSTMCell gate order i,f,g,o over z[M,4*FM]; updates c,h in place.
__global__ void lstm_k(const float* __restrict__ Z, float* __restrict__ cst,
                       float* __restrict__ hst, int M) {
  int idx = blockIdx.x * blockDim.x + threadIdx.x;
  if (idx >= M * FM) return;
  int row = idx >> 8, f = idx & (FM - 1);
  const float* z = Z + (size_t)row * (4 * FM);
  float ig = sigmoidf_(z[f]);
  float fg = sigmoidf_(z[FM + f]);
  float g  = tanhf(z[2 * FM + f]);
  float og = sigmoidf_(z[3 * FM + f]);
  float cn = fg * cst[idx] + ig * g;
  cst[idx] = cn;
  hst[idx] = og * tanhf(cn);
}

// logits[v] = H[v,:512] . W2 + b   (one wave32 per row)
__global__ __launch_bounds__(256) void vote_dot_k(const float* __restrict__ H,
                                                  const float* __restrict__ W,
                                                  const float* __restrict__ b,
                                                  float* __restrict__ out) {
  int lane = threadIdx.x & 31;
  int row = blockIdx.x * 8 + (threadIdx.x >> 5);
  if (row >= NVARS) return;
  const float* h = H + (size_t)row * (2 * FM);
  float s = 0.0f;
  for (int k = lane; k < 2 * FM; k += 32) s += h[k] * W[k];
#pragma unroll
  for (int off = 16; off > 0; off >>= 1) s += __shfl_down(s, off);
  if (lane == 0) out[row] = s + b[0];
}

__device__ __forceinline__ float softplusf_(float x) {
  return (x > 0.0f) ? (x + log1pf(expf(-x))) : log1pf(expf(x));
}

__global__ void clause_loss_k(const float* __restrict__ logits,
                              const int* __restrict__ varIdx,
                              const float* __restrict__ sgn,
                              float* __restrict__ loss) {
  int c = blockIdx.x * blockDim.x + threadIdx.x;
  if (c >= NCLS) return;
  float s = 0.0f;
#pragma unroll
  for (int j = 0; j < 3; ++j) {
    int e = 3 * c + j;
    s += softplusf_(logits[varIdx[e]] * sgn[e]);
  }
  float cv = expf(-s);
  float t = -logf(1.0f - cv + 1e-8f);
  atomicAdd(loss, t * t);
}

__global__ void finalize_k(const float* __restrict__ logits,
                           const float* __restrict__ loss,
                           float* __restrict__ out) {
  int i = blockIdx.x * blockDim.x + threadIdx.x;
  if (i > NVARS) return;
  if (i < NVARS) out[i] = logits[i];
  else           out[NVARS] = loss[0] * (1.0f / (float)ROUNDS);
}

// ---------------------------------------------------------------------------
// Host orchestration
// ---------------------------------------------------------------------------
#define CDIV(a, b) (((a) + (b)-1) / (b))

extern "C" void kernel_launch(void* const* d_in, const int* in_sizes, int n_in,
                              void* d_out, int out_size, void* d_ws,
                              size_t ws_size, hipStream_t stream) {
  (void)in_sizes; (void)n_in; (void)out_size; (void)ws_size;

  const int*   clause_lits = (const int*)d_in[0];
  const float* L_init = (const float*)d_in[2];
  const float* C_init = (const float*)d_in[3];
  const float* LC_W0 = (const float*)d_in[4];  const float* LC_b0 = (const float*)d_in[5];
  const float* LC_W1 = (const float*)d_in[6];  const float* LC_b1 = (const float*)d_in[7];
  const float* LC_W2 = (const float*)d_in[8];  const float* LC_b2 = (const float*)d_in[9];
  const float* CL_W0 = (const float*)d_in[10]; const float* CL_b0 = (const float*)d_in[11];
  const float* CL_W1 = (const float*)d_in[12]; const float* CL_b1 = (const float*)d_in[13];
  const float* CL_W2 = (const float*)d_in[14]; const float* CL_b2 = (const float*)d_in[15];
  const float* C_Wx = (const float*)d_in[16];  const float* C_Wh = (const float*)d_in[17];
  const float* C_b  = (const float*)d_in[18];
  const float* L_Wx = (const float*)d_in[19];  const float* L_Wh = (const float*)d_in[20];
  const float* L_b  = (const float*)d_in[21];
  const float* V_W0 = (const float*)d_in[22];  const float* V_b0 = (const float*)d_in[23];
  const float* V_W1 = (const float*)d_in[24];  const float* V_b1 = (const float*)d_in[25];
  const float* V_W2 = (const float*)d_in[26];  const float* V_b2 = (const float*)d_in[27];

  // workspace allocator (256B aligned)
  size_t off = 0;
  char* base = (char*)d_ws;
  auto alloc = [&](size_t bytes) -> void* {
    size_t a = (off + 255) & ~(size_t)255;
    off = a + bytes;
    return (void*)(base + a);
  };

  float* lh   = (float*)alloc((size_t)NLITS * FM * 4);
  float* lcs  = (float*)alloc((size_t)NLITS * FM * 4);
  float* chh  = (float*)alloc((size_t)NCLS * FM * 4);
  float* ccs  = (float*)alloc((size_t)NCLS * FM * 4);
  float* pre  = (float*)alloc((size_t)NCLS * FM * 4);   // MLP outputs (max rows)
  float* t1   = (float*)alloc((size_t)NCLS * FM * 4);   // also vote [4000,512]
  float* t2   = (float*)alloc((size_t)NCLS * FM * 4);
  float* msgs = (float*)alloc((size_t)NCLS * FM * 4);
  float* clm  = (float*)alloc((size_t)NLITS * FM * 4);
  float* xcat = (float*)alloc((size_t)NLITS * 2 * FM * 4);
  float* zbuf = (float*)alloc((size_t)NCLS * 4 * FM * 4);
  float* logits = (float*)alloc((size_t)NVARS * 4);
  float* lossB  = (float*)alloc(256);
  int*   litIdx = (int*)alloc((size_t)NEDGE * 4);
  int*   varIdx = (int*)alloc((size_t)NEDGE * 4);
  float* sgn    = (float*)alloc((size_t)NEDGE * 4);

  auto packW = [&](const float* W, int Kd, int N) -> _Float16* {
    _Float16* P = (_Float16*)alloc((size_t)Kd * N * 2);
    int total = (Kd >> 5) * (N >> 4) * 32;
    pack_w_k<<<CDIV(total, 256), 256, 0, stream>>>(W, P, Kd, N);
    return P;
  };

  _Float16* pLC0 = packW(LC_W0, FM, FM);
  _Float16* pLC1 = packW(LC_W1, FM, FM);
  _Float16* pLC2 = packW(LC_W2, FM, FM);
  _Float16* pCL0 = packW(CL_W0, FM, FM);
  _Float16* pCL1 = packW(CL_W1, FM, FM);
  _Float16* pCL2 = packW(CL_W2, FM, FM);
  _Float16* pCWx = packW(C_Wx, FM, 4 * FM);
  _Float16* pCWh = packW(C_Wh, FM, 4 * FM);
  _Float16* pLWx = packW(L_Wx, 2 * FM, 4 * FM);
  _Float16* pLWh = packW(L_Wh, FM, 4 * FM);
  _Float16* pV0  = packW(V_W0, 2 * FM, 2 * FM);
  _Float16* pV1  = packW(V_W1, 2 * FM, 2 * FM);

  auto gemm = [&](const float* A, const _Float16* Bp, const float* bias,
                  const float* prev, float* C, int M, int N, int Kd, int relu) {
    dim3 g(N / 64, CDIV(M, 128));
    gemm_wmma_k<<<g, 256, 0, stream>>>(A, Bp, bias, prev, C, M, N, Kd, relu);
  };

  // init
  edges_k<<<CDIV(NEDGE, 256), 256, 0, stream>>>(clause_lits, litIdx, varIdx, sgn);
  init_state_k<<<CDIV(NLITS * FM, 256), 256, 0, stream>>>(lh, lcs, L_init, NLITS);
  init_state_k<<<CDIV(NCLS * FM, 256), 256, 0, stream>>>(chh, ccs, C_init, NCLS);
  fill_k<<<1, 32, 0, stream>>>(lossB, 0.0f, 1);

  for (int r = 0; r < ROUNDS; ++r) {
    // --- literal -> clause messages: LC MLP3 + segment sum
    gemm(lh, pLC0, LC_b0, nullptr, t1, NLITS, FM, FM, 1);
    gemm(t1, pLC1, LC_b1, nullptr, t2, NLITS, FM, FM, 1);
    gemm(t2, pLC2, LC_b2, nullptr, pre, NLITS, FM, FM, 0);
    gather3_k<<<CDIV(NCLS * 64, 256), 256, 0, stream>>>(pre, litIdx, msgs);

    // --- clause LSTM: z = msgs@Wx + b ; z += ch@Wh ; elementwise
    gemm(msgs, pCWx, C_b, nullptr, zbuf, NCLS, 4 * FM, FM, 0);
    gemm(chh,  pCWh, nullptr, zbuf, zbuf, NCLS, 4 * FM, FM, 0);
    lstm_k<<<CDIV(NCLS * FM, 256), 256, 0, stream>>>(zbuf, ccs, chh, NCLS);

    // --- clause -> literal messages: CL MLP3 + scatter
    gemm(chh, pCL0, CL_b0, nullptr, t1, NCLS, FM, FM, 1);
    gemm(t1,  pCL1, CL_b1, nullptr, t2, NCLS, FM, FM, 1);
    gemm(t2,  pCL2, CL_b2, nullptr, pre, NCLS, FM, FM, 0);
    fill_k<<<CDIV(NLITS * FM, 256), 256, 0, stream>>>(clm, 0.0f, NLITS * FM);
    scatter_add_k<<<CDIV(NEDGE * 64, 256), 256, 0, stream>>>(pre, litIdx, clm);

    // --- literal LSTM on concat([cl_msgs, flipped(lh)])
    build_xcat_k<<<CDIV(NLITS * 2 * FM, 256), 256, 0, stream>>>(clm, lh, xcat);
    gemm(xcat, pLWx, L_b, nullptr, zbuf, NLITS, 4 * FM, 2 * FM, 0);
    gemm(lh,   pLWh, nullptr, zbuf, zbuf, NLITS, 4 * FM, FM, 0);
    lstm_k<<<CDIV(NLITS * FM, 256), 256, 0, stream>>>(zbuf, lcs, lh, NLITS);

    // --- vote + clause loss
    build_vcat_k<<<CDIV(NVARS * 2 * FM, 256), 256, 0, stream>>>(lh, xcat);
    gemm(xcat, pV0, V_b0, nullptr, t1, NVARS, 2 * FM, 2 * FM, 1);
    gemm(t1,   pV1, V_b1, nullptr, t2, NVARS, 2 * FM, 2 * FM, 1);
    vote_dot_k<<<CDIV(NVARS, 8), 256, 0, stream>>>(t2, V_W2, V_b2, logits);
    clause_loss_k<<<CDIV(NCLS, 256), 256, 0, stream>>>(logits, varIdx, sgn, lossB);
  }

  // outputs: logits (final round's vote == vote(final lh)) then loss/ROUNDS
  finalize_k<<<CDIV(NVARS + 1, 256), 256, 0, stream>>>(logits, lossB,
                                                       (float*)d_out);
}
